// TalkingHeadsAttention_48670569398890
// MI455X (gfx1250) — matrix operand order repaired and checked by reading, not compiled, not run on hardware
//
#include <hip/hip_runtime.h>

// ---------------- CDNA5 WMMA types ----------------
typedef __attribute__((ext_vector_type(16))) __bf16 v16bf;
typedef __attribute__((ext_vector_type(8)))  float  v8f;

#define WMMA_BF16(a, b, c) \
  __builtin_amdgcn_wmma_f32_16x16x32_bf16(false, (a), false, (b), (short)0, (c), false, false)

// Problem constants
#define BB 4
#define SS 1024
#define DD 1024
#define NN 16
#define HH 64

// =====================================================================
// Pack kernels: rearrange operands into WMMA-ready lane-major tiles.
// A-tile (16x32, MxK, bf16): lane = m + 16*g; slot j,p -> k = (j>=4?16:0)+g*8+(j&3)*2+p
// B-tile (32x16, KxN, bf16): lane = n + 16*g; slot j,p -> k = (j>=4?16:0)+g*8+(j&3)*2+p
// Each tile = 32 lanes * 16 halves = 1 KB, stored contiguously.
// =====================================================================
template <typename SrcT>
__global__ void pack_A_kernel(const SrcT* __restrict__ src, __bf16* __restrict__ dst,
                              int tilesM, int tilesK, long rowStride,
                              long outerStride, long innerStride, int nInner, float scale) {
  int wave = blockIdx.x * 8 + (threadIdx.x >> 5);
  int lane = threadIdx.x & 31;
  int perBatch = tilesM * tilesK;
  int batch = wave / perBatch;
  int r     = wave % perBatch;
  int tileM = r / tilesK, tileK = r % tilesK;
  const SrcT* sb = src + (long)(batch / nInner) * outerStride + (long)(batch % nInner) * innerStride;
  __bf16* db = dst + ((size_t)batch * perBatch + r) * 512;
  int m = tileM * 16 + (lane & 15);
  int g = lane >> 4;
  v16bf outv;
#pragma unroll
  for (int j = 0; j < 8; ++j)
#pragma unroll
    for (int p = 0; p < 2; ++p) {
      int k = tileK * 32 + ((j >= 4) ? 16 : 0) + g * 8 + (j & 3) * 2 + p;
      float v = (float)sb[(long)m * rowStride + k] * scale;
      outv[j * 2 + p] = (__bf16)v;
    }
  *(v16bf*)(db + lane * 16) = outv;
}

template <typename SrcT>
__global__ void pack_B_kernel(const SrcT* __restrict__ src, __bf16* __restrict__ dst,
                              int tilesK, int tilesC, long kStride, long cStride,
                              long outerStride, long innerStride, int nInner) {
  int wave = blockIdx.x * 8 + (threadIdx.x >> 5);
  int lane = threadIdx.x & 31;
  int perBatch = tilesK * tilesC;
  int batch = wave / perBatch;
  int r     = wave % perBatch;
  int tileK = r / tilesC, tileC = r % tilesC;
  const SrcT* sb = src + (long)(batch / nInner) * outerStride + (long)(batch % nInner) * innerStride;
  __bf16* db = dst + ((size_t)batch * perBatch + r) * 512;
  int col = tileC * 16 + (lane & 15);
  int g = lane >> 4;
  v16bf outv;
#pragma unroll
  for (int j = 0; j < 8; ++j)
#pragma unroll
    for (int p = 0; p < 2; ++p) {
      int k = tileK * 32 + ((j >= 4) ? 16 : 0) + g * 8 + (j & 3) * 2 + p;
      float v = (float)sb[(long)k * kStride + (long)col * cStride];
      outv[j * 2 + p] = (__bf16)v;
    }
  *(v16bf*)(db + lane * 16) = outv;
}

// =====================================================================
// Generic packed-operand WMMA GEMM: out[M x C] = A[M x K] * B[K x C] + bias[C]
// Block = 256 threads = 8 waves (2 M-tiles x 4 C-groups); each wave -> 16x64 output.
// =====================================================================
template <typename OutT>
__global__ void gemm_bf16_kernel(const __bf16* __restrict__ A, const __bf16* __restrict__ Bm,
                                 const float* __restrict__ bias, OutT* __restrict__ out,
                                 int M, int C, int K) {
  int lane = threadIdx.x & 31;
  int w = threadIdx.x >> 5;
  int wm = w >> 2, wc = w & 3;
  int tilesK = K >> 5, tilesC = C >> 4;
  int mTile = blockIdx.y * 2 + wm;
  int cTile0 = blockIdx.x * 16 + wc * 4;
  const v16bf* Av = (const v16bf*)A;
  const v16bf* Bv = (const v16bf*)Bm;
  v8f acc[4] = {};
  for (int kt = 0; kt < tilesK; ++kt) {
    v16bf a = Av[(size_t)(mTile * tilesK + kt) * 32 + lane];
#pragma unroll
    for (int c = 0; c < 4; ++c) {
      v16bf b = Bv[(size_t)(kt * tilesC + cTile0 + c) * 32 + lane];
      acc[c] = WMMA_BF16(a, b, acc[c]);
    }
  }
  int g = lane >> 4, cl = lane & 15;
#pragma unroll
  for (int c = 0; c < 4; ++c)
#pragma unroll
    for (int j = 0; j < 8; ++j) {
      int row = mTile * 16 + j + 8 * g;
      int col = (cTile0 + c) * 16 + cl;
      float v = acc[c][j] + bias[col];
      out[(size_t)row * C + col] = (OutT)v;
    }
}

// =====================================================================
// Scores + pre-softmax head mix + mask.
// Block (b, fTile, tBlock of 256 cols): 8 waves, wave w computes heads 2w,2w+1
// via WMMA (K=64 -> two 16x16x32 bf16 ops), tiles staged in LDS, then a VALU
// 16x16 head-mix per (f,t), additive mask, f32 store of scores[b,l,f,t].
// =====================================================================
__global__ void scores_premix_kernel(const __bf16* __restrict__ qA, const __bf16* __restrict__ kB,
                                     const float* __restrict__ preW, const int* __restrict__ mask,
                                     float* __restrict__ scores) {
  __shared__ float sT[16][16][16];
  __shared__ float Wl[256];
  int b = blockIdx.z, fTile = blockIdx.y, tBlock = blockIdx.x;
  int tid = threadIdx.x, lane = tid & 31, w = tid >> 5;
  Wl[tid] = preW[tid];
  const v16bf* qv = (const v16bf*)qA;
  const v16bf* kv = (const v16bf*)kB;
  int n0 = w * 2;
  v16bf qa[2][2];
#pragma unroll
  for (int h = 0; h < 2; ++h)
#pragma unroll
    for (int kt = 0; kt < 2; ++kt) {
      size_t batch = (size_t)(b * 16 + n0 + h);
      qa[h][kt] = qv[(batch * 128 + (size_t)fTile * 2 + kt) * 32 + lane];
    }
  __syncthreads();
  for (int tt = 0; tt < 16; ++tt) {
    int tTile = tBlock * 16 + tt;
#pragma unroll
    for (int h = 0; h < 2; ++h) {
      int n = n0 + h;
      size_t batch = (size_t)(b * 16 + n);
      v16bf b0 = kv[(batch * 128 + 0 * 64 + tTile) * 32 + lane];
      v16bf b1 = kv[(batch * 128 + 64 + tTile) * 32 + lane];
      v8f c = {};
      c = WMMA_BF16(qa[h][0], b0, c);
      c = WMMA_BF16(qa[h][1], b1, c);
      int g = lane >> 4, cl = lane & 15;
#pragma unroll
      for (int j = 0; j < 8; ++j) sT[n][j + 8 * g][cl] = c[j];
    }
    __syncthreads();
    {
      int f = tid >> 4, t = tid & 15;
      int fg = fTile * 16 + f, tg = tTile * 16 + t;
      float mixed[16];
#pragma unroll
      for (int l = 0; l < 16; ++l) mixed[l] = 0.0f;
#pragma unroll
      for (int n = 0; n < 16; ++n) {
        float s = sT[n][f][t];
#pragma unroll
        for (int l = 0; l < 16; ++l) mixed[l] += s * Wl[n * 16 + l];
      }
      float madd = (1.0f - (float)mask[((size_t)b * SS + fg) * SS + tg]) * -10000.0f;
#pragma unroll
      for (int l = 0; l < 16; ++l)
        scores[(((size_t)b * 16 + l) * SS + fg) * SS + tg] = mixed[l] + madd;
    }
    __syncthreads();
  }
}

// =====================================================================
// Softmax over t for all 16 mixed heads of one (b,f) row, then post-softmax
// head mix, writing mixed probs DIRECTLY in packed WMMA-A layout (bf16).
// =====================================================================
__global__ void softmax_postmix_kernel(const float* __restrict__ scores,
                                       const float* __restrict__ postW,
                                       __bf16* __restrict__ probsA) {
  __shared__ float rows[16][1024 + 8];
  __shared__ float red[16][16];
  __shared__ float maxv[16];
  __shared__ float invs[16];
  __shared__ float Wl[256];
  int b = blockIdx.y, f = blockIdx.x, tid = threadIdx.x;
  Wl[tid] = postW[tid];
  for (int l = 0; l < 16; ++l)
    for (int i = tid; i < 1024; i += 256)
      rows[l][i] = scores[(((size_t)b * 16 + l) * SS + f) * SS + i];
  __syncthreads();
  int l = tid >> 4, sub = tid & 15;
  float m = -3.4e38f;
  for (int i = 0; i < 64; ++i) m = fmaxf(m, rows[l][sub * 64 + i]);
  red[l][sub] = m;
  __syncthreads();
  if (tid < 16) {
    float mm = red[tid][0];
    for (int i = 1; i < 16; ++i) mm = fmaxf(mm, red[tid][i]);
    maxv[tid] = mm;
  }
  __syncthreads();
  float mx = maxv[l];
  float s = 0.0f;
  for (int i = 0; i < 64; ++i) s += __expf(rows[l][sub * 64 + i] - mx);
  red[l][sub] = s;
  __syncthreads();
  if (tid < 16) {
    float tot = 0.0f;
    for (int i = 0; i < 16; ++i) tot += red[tid][i];
    invs[tid] = 1.0f / tot;
  }
  __syncthreads();
  float inv = invs[l];
  for (int i = 0; i < 64; ++i) {
    int x = sub * 64 + i;
    rows[l][x] = __expf(rows[l][x] - mx) * inv;
  }
  __syncthreads();
  // post-softmax mix + packed-A store: thread = (head n, t-lane)
  int n = tid >> 4, tsub = tid & 15;
  int fTile = f >> 4, mrow = f & 15;
  size_t batchBase = ((size_t)b * 16 + n) * (size_t)(64 * 32) * 512;
  for (int i = 0; i < 64; ++i) {
    int t = tsub + i * 16;
    float acc = 0.0f;
#pragma unroll
    for (int ll = 0; ll < 16; ++ll) acc += rows[ll][t] * Wl[ll * 16 + n];
    int tTile = t >> 5, k = t & 31;
    int g = (k >> 3) & 1;
    int j = ((k & 16) ? 4 : 0) + ((k & 7) >> 1);
    int p = k & 1;
    int lane_ = mrow + (g << 4);
    probsA[batchBase + (size_t)(fTile * 32 + tTile) * 512 + lane_ * 16 + j * 2 + p] = (__bf16)acc;
  }
}

// =====================================================================
// ctx[b,f,n,h] = sum_t probs_mixed[b,n,f,t] * v[b,t,n,h]  (WMMA, K over t)
// Block (fTile, b): 8 waves x 2 heads; per head 16(f) x 64(h) output, K=1024.
// =====================================================================
__global__ void ctx_gemm_kernel(const __bf16* __restrict__ probsA, const __bf16* __restrict__ vB,
                                __bf16* __restrict__ ctx) {
  int b = blockIdx.y, fTile = blockIdx.x;
  int tid = threadIdx.x, lane = tid & 31, w = tid >> 5;
  const v16bf* Av = (const v16bf*)probsA;
  const v16bf* Bv = (const v16bf*)vB;
#pragma unroll
  for (int h2 = 0; h2 < 2; ++h2) {
    int n = w * 2 + h2;
    size_t aBase = (((size_t)b * 16 + n) * 2048 + (size_t)fTile * 32) * 32;
    size_t bBase = ((size_t)b * 16 + n) * 128 * 32;
    v8f acc[4] = {};
    for (int kt = 0; kt < 32; ++kt) {
      v16bf a = Av[aBase + (size_t)kt * 32 + lane];
#pragma unroll
      for (int c = 0; c < 4; ++c) {
        v16bf bb = Bv[bBase + (size_t)(kt * 4 + c) * 32 + lane];
        acc[c] = WMMA_BF16(a, bb, acc[c]);
      }
    }
    int g = lane >> 4, cl = lane & 15;
#pragma unroll
    for (int c = 0; c < 4; ++c)
#pragma unroll
      for (int j = 0; j < 8; ++j) {
        int fr = fTile * 16 + j + 8 * g;
        int h = c * 16 + cl;
        ctx[(((size_t)b * SS + fr) * 16 + n) * 64 + h] = (__bf16)acc[c][j];
      }
  }
}

// =====================================================================
extern "C" void kernel_launch(void* const* d_in, const int* in_sizes, int n_in,
                              void* d_out, int out_size, void* d_ws, size_t ws_size,
                              hipStream_t stream) {
  const float* from  = (const float*)d_in[0];
  const float* to    = (const float*)d_in[1];
  const int*   mask  = (const int*)d_in[2];
  const float* wq    = (const float*)d_in[3];
  const float* bq    = (const float*)d_in[4];
  const float* wk    = (const float*)d_in[5];
  const float* bk    = (const float*)d_in[6];
  const float* wv    = (const float*)d_in[7];
  const float* bv    = (const float*)d_in[8];
  const float* preW  = (const float*)d_in[9];
  const float* postW = (const float*)d_in[10];
  const float* wo    = (const float*)d_in[11];
  const float* bo    = (const float*)d_in[12];
  float* out = (float*)d_out;

  char* ws = (char*)d_ws;
  size_t off = 0;
  auto alloc = [&](size_t bytes) -> char* {
    char* p = ws + off;
    off = (off + bytes + 255) & ~(size_t)255;
    return p;
  };
  const size_t M = (size_t)BB * SS;           // 4096
  __bf16* fromA = (__bf16*)alloc(M * DD * 2);           // packed A of from_tensor
  __bf16* toA   = (__bf16*)alloc(M * DD * 2);           // packed A of to_tensor
  __bf16* wqB   = (__bf16*)alloc((size_t)DD * DD * 2);
  __bf16* wkB   = (__bf16*)alloc((size_t)DD * DD * 2);
  __bf16* wvB   = (__bf16*)alloc((size_t)DD * DD * 2);
  __bf16* woB   = (__bf16*)alloc((size_t)DD * DD * 2);
  __bf16* qb    = (__bf16*)alloc(M * DD * 2);           // [b,s,n,h]
  __bf16* kb    = (__bf16*)alloc(M * DD * 2);
  __bf16* vb    = (__bf16*)alloc(M * DD * 2);
  __bf16* qA    = (__bf16*)alloc(M * DD * 2);           // per (b,n) packed A over (f,h)
  __bf16* kBp   = (__bf16*)alloc(M * DD * 2);           // per (b,n) packed B over (h,t)
  __bf16* vBp   = (__bf16*)alloc(M * DD * 2);           // per (b,n) packed B over (t,h)
  float*  sc    = (float*)alloc((size_t)BB * NN * SS * SS * 4);   // mixed scores f32
  __bf16* prA   = (__bf16*)alloc((size_t)BB * NN * SS * SS * 2);  // mixed probs, packed A
  __bf16* ctx   = (__bf16*)alloc(M * DD * 2);           // [b,f,n,h]
  __bf16* ctxA  = (__bf16*)alloc(M * DD * 2);
  (void)ws_size; (void)in_sizes; (void)n_in; (void)out_size;

  const long bnOuter = (long)SS * NN * HH;  // per-b stride in qkv [b,s,n,h]
  dim3 blk(256);

  // --- pack activations (f32 -> bf16 packed A), M=4096 K=1024 ---
  pack_A_kernel<float><<<1024, blk, 0, stream>>>(from, fromA, 256, 32, DD, 0, 0, 1, 1.0f);
  pack_A_kernel<float><<<1024, blk, 0, stream>>>(to,   toA,   256, 32, DD, 0, 0, 1, 1.0f);
  // --- pack weights (f32 -> bf16 packed B), K=1024 C=1024 ---
  pack_B_kernel<float><<<256, blk, 0, stream>>>(wq, wqB, 32, 64, DD, 1, 0, 0, 1);
  pack_B_kernel<float><<<256, blk, 0, stream>>>(wk, wkB, 32, 64, DD, 1, 0, 0, 1);
  pack_B_kernel<float><<<256, blk, 0, stream>>>(wv, wvB, 32, 64, DD, 1, 0, 0, 1);
  pack_B_kernel<float><<<256, blk, 0, stream>>>(wo, woB, 32, 64, DD, 1, 0, 0, 1);

  // --- QKV projections (WMMA GEMM, bf16 out [b,s,n,h]) ---
  dim3 ggrid(DD / 256, M / 32);
  gemm_bf16_kernel<__bf16><<<ggrid, blk, 0, stream>>>(fromA, wqB, bq, qb, (int)M, DD, DD);
  gemm_bf16_kernel<__bf16><<<ggrid, blk, 0, stream>>>(toA,   wkB, bk, kb, (int)M, DD, DD);
  gemm_bf16_kernel<__bf16><<<ggrid, blk, 0, stream>>>(toA,   wvB, bv, vb, (int)M, DD, DD);

  // --- repack per (b,n) for attention GEMMs; 1/sqrt(H)=0.125 folded into q ---
  pack_A_kernel<__bf16><<<1024, blk, 0, stream>>>(qb, qA, 64, 2, NN * HH, bnOuter, HH, NN, 0.125f);
  pack_B_kernel<__bf16><<<1024, blk, 0, stream>>>(kb, kBp, 2, 64, 1, NN * HH, bnOuter, HH, NN);
  pack_B_kernel<__bf16><<<1024, blk, 0, stream>>>(vb, vBp, 32, 4, NN * HH, 1, bnOuter, HH, NN);

  // --- scores = q k^T (WMMA) -> pre-softmax head mix -> +mask, f32 [b,l,f,t] ---
  scores_premix_kernel<<<dim3(4, 64, BB), blk, 0, stream>>>(qA, kBp, preW, mask, sc);

  // --- softmax over t + post-softmax head mix -> packed-A bf16 probs ---
  softmax_postmix_kernel<<<dim3(SS, BB), blk, 0, stream>>>(sc, postW, prA);

  // --- ctx = probs_mixed * v (WMMA) -> bf16 [b,f,n,h] ---
  ctx_gemm_kernel<<<dim3(64, BB), blk, 0, stream>>>(prA, vBp, ctx);

  // --- output projection: repack ctx, GEMM with wo + bo -> f32 d_out ---
  pack_A_kernel<__bf16><<<1024, blk, 0, stream>>>(ctx, ctxA, 256, 32, DD, 0, 0, 1, 1.0f);
  gemm_bf16_kernel<float><<<ggrid, blk, 0, stream>>>(ctxA, woB, bo, out, (int)M, DD, DD);
}
